// PNI_mamba_48455821033747
// MI455X (gfx1250) — matrix operand order repaired and checked by reading, compiled, not run on hardware
//
#include <hip/hip_runtime.h>
#include <hip/hip_bf16.h>
#include <math.h>

#define D_MODEL 256
#define D_STATE 16
#define D_CONVK 4
#define D_INNER 512
#define DT_RANK 16
#define BATCH   16
#define LPEP    512
#define LLIG    64

typedef __attribute__((ext_vector_type(16))) __bf16 v16bf;
typedef __attribute__((ext_vector_type(8)))  float  v8f;

__device__ __forceinline__ float siluf(float x) { return x / (1.f + __expf(-x)); }
__device__ __forceinline__ float softplusf(float x) {
  return (x > 20.f) ? x : log1pf(__expf(x));
}

// ---------------------------------------------------------------------------
// WMMA GEMM: C[M,N] = act(A[M,K] @ W[N,K]^T + bias[N])
// fp32 in memory, bf16 WMMA compute (v_wmma_f32_16x16x32_bf16).
//
// Requirements (guaranteed by all call sites):
//   M % 16 == 0, N % 16 == 0.  K may be arbitrary (tail chunk handles it);
//   if K >= 32 then lda % 4 == 0 (float4-aligned main-loop loads).
//
// Each wave computes a 16(M) x 64(N) tile: one A fragment per k-step feeds
// four v_wmma into four accumulators.  4 waves per block tile 256 along N.
//
// Fragment layouts per CDNA5 ISA 7.12.2 (wave32):
//   A 16x32 bf16: lane&15 = M row; k = k0 + (i&7) + ((i&8)?16:0), k0=(lane&16)?8:0
//   B 32x16 bf16: lane&15 = N column; element i has the same k as A
//   C/D 16x16 f32: lane&15 = N; VGPR j -> M = j + ((lane&16)?8:0)
// act: 0 = none, 1 = softplus
// ---------------------------------------------------------------------------
__global__ void PNI_gemm_wmma(const float* __restrict__ A, int lda,
                              const float* __restrict__ W,
                              const float* __restrict__ bias,
                              float* __restrict__ C,
                              int M, int N, int K, int act)
{
  const int lane  = threadIdx.x & 31;
  const int wave  = threadIdx.x >> 5;
  const int tm    = blockIdx.x;                 // 16-row M tile
  const int nBase = (blockIdx.y * 4 + wave) * 64;
  if (nBase >= N) return;
  const int nsub = min(4, (N - nBase) >> 4);    // active 16-wide N subtiles

  const int row = tm * 16 + (lane & 15);
  const int k0  = (lane & 16) ? 8 : 0;
  const float* __restrict__ arow = A + (long)row * lda;

  v8f c[4];
#pragma unroll
  for (int nt = 0; nt < 4; ++nt)
    c[nt] = (v8f){0.f, 0.f, 0.f, 0.f, 0.f, 0.f, 0.f, 0.f};

  const int K32 = K & ~31;

  // ---- main K loop: no bounds checks, vectorized A loads -----------------
  for (int kt = 0; kt < K32; kt += 32) {
    v16bf a;
    {
      const float4 a0 = *(const float4*)(arow + kt + k0);
      const float4 a1 = *(const float4*)(arow + kt + k0 + 4);
      const float4 a2 = *(const float4*)(arow + kt + k0 + 16);
      const float4 a3 = *(const float4*)(arow + kt + k0 + 20);
      a[0]=(__bf16)a0.x;  a[1]=(__bf16)a0.y;  a[2]=(__bf16)a0.z;  a[3]=(__bf16)a0.w;
      a[4]=(__bf16)a1.x;  a[5]=(__bf16)a1.y;  a[6]=(__bf16)a1.z;  a[7]=(__bf16)a1.w;
      a[8]=(__bf16)a2.x;  a[9]=(__bf16)a2.y;  a[10]=(__bf16)a2.z; a[11]=(__bf16)a2.w;
      a[12]=(__bf16)a3.x; a[13]=(__bf16)a3.y; a[14]=(__bf16)a3.z; a[15]=(__bf16)a3.w;
    }
    const long kb = kt + k0;                    // B K base for this lane half
    for (int nt = 0; nt < nsub; ++nt) {
      const float* __restrict__ wp = W + (long)(nBase + nt * 16 + (lane & 15)) * K + kb;
      v16bf b;
#pragma unroll
      for (int i = 0; i < 16; ++i)              // lane&15 picks the W row (lanes K apart)
        b[i] = (__bf16)wp[(i & 7) + ((i & 8) ? 16 : 0)];
      c[nt] = __builtin_amdgcn_wmma_f32_16x16x32_bf16(false, a, false, b,
                                                      (short)0, c[nt], false, false);
    }
  }

  // ---- K tail (K % 32 != 0): clamp-address + select-zero, branch-free ----
  if (K32 < K) {
    const int kt = K32;
    v16bf a;
#pragma unroll
    for (int i = 0; i < 16; ++i) {
      int k = kt + k0 + (i & 7) + ((i & 8) ? 16 : 0);
      int kc = (k < K) ? k : (K - 1);
      float v = arow[kc];
      if (k >= K) v = 0.f;
      a[i] = (__bf16)v;
    }
    const int kb  = kt + k0;
    for (int nt = 0; nt < nsub; ++nt) {
      const float* __restrict__ wp = W + (long)(nBase + nt * 16 + (lane & 15)) * K;
      v16bf b;
#pragma unroll
      for (int i = 0; i < 16; ++i) {
        int k = kb + (i & 7) + ((i & 8) ? 16 : 0);
        int kc = (k < K) ? k : (K - 1);
        float v = wp[kc];
        if (k >= K) v = 0.f;
        b[i] = (__bf16)v;
      }
      c[nt] = __builtin_amdgcn_wmma_f32_16x16x32_bf16(false, a, false, b,
                                                      (short)0, c[nt], false, false);
    }
  }

  // ---- epilogue: bias + activation + store (no element guards needed) ----
  const int m0 = tm * 16 + ((lane & 16) ? 8 : 0);
  for (int nt = 0; nt < nsub; ++nt) {
    const int n  = nBase + nt * 16 + (lane & 15);
    const float bv = bias ? bias[n] : 0.f;
#pragma unroll
    for (int j = 0; j < 8; ++j) {
      float v = c[nt][j] + bv;
      if (act == 1) v = softplusf(v);
      C[(long)(m0 + j) * N + n] = v;
    }
  }
}

// ---------------------------------------------------------------------------
// Causal depthwise conv (width 4) + bias + SiLU.  xz is (B,L,2*D_INNER),
// the x half is channels [0,512).  Output xc is (B,L,D_INNER).
// ---------------------------------------------------------------------------
__global__ void PNI_conv_silu(const float* __restrict__ xz,
                              const float* __restrict__ cw,
                              const float* __restrict__ cb,
                              float* __restrict__ xc, int L)
{
  const int idx = blockIdx.x * blockDim.x + threadIdx.x;
  const int total = BATCH * L * D_INNER;
  if (idx >= total) return;
  const int d = idx % D_INNER;
  const int l = (idx / D_INNER) % L;
  const int b = idx / (D_INNER * L);
  float acc = cb[d];
#pragma unroll
  for (int j = 0; j < D_CONVK; ++j) {
    int ls = l - (D_CONVK - 1) + j;
    if (ls >= 0)
      acc += cw[d * D_CONVK + j] *
             xz[((long)(b * L + ls)) * (2 * D_INNER) + d];
  }
  xc[idx] = siluf(acc);
}

// ---------------------------------------------------------------------------
// Fused selective scan. One block per batch, one thread per channel d.
// State h[16] in registers, Bs/Cs staged in LDS per timestep, dA computed
// on the fly (avoids materializing the 67M-element dA/dBu tensors).
// Fuses the D-skip connection and the silu(z) gate; y written in place.
// ---------------------------------------------------------------------------
__global__ void PNI_scan(const float* __restrict__ dbl,
                         const float* __restrict__ dt,
                         const float* __restrict__ xc_in,
                         const float* __restrict__ xz,
                         const float* __restrict__ A_log,
                         const float* __restrict__ Dp,
                         float* __restrict__ y, int L)
{
  __shared__ float sB[D_STATE];
  __shared__ float sC[D_STATE];
  const int b = blockIdx.x;
  const int d = threadIdx.x;

  float Arow[D_STATE];
#pragma unroll
  for (int s = 0; s < D_STATE; ++s)
    Arow[s] = -__expf(A_log[d * D_STATE + s]);
  const float Dv = Dp[d];
  float h[D_STATE];
#pragma unroll
  for (int s = 0; s < D_STATE; ++s) h[s] = 0.f;

  for (int l = 0; l < L; ++l) {
    const long rl = (long)(b * L + l);
    if (threadIdx.x < 2 * D_STATE) {
      const int t = threadIdx.x;
      float v = dbl[rl * (DT_RANK + 2 * D_STATE) + DT_RANK + t];
      if (t < D_STATE) sB[t] = v; else sC[t - D_STATE] = v;
    }
    __syncthreads();
    const float dtv = dt[rl * D_INNER + d];
    const float xv  = xc_in[rl * D_INNER + d];
    float acc = 0.f;
#pragma unroll
    for (int s = 0; s < D_STATE; ++s) {
      float dA = __expf(dtv * Arow[s]);
      h[s] = dA * h[s] + dtv * sB[s] * xv;
      acc += h[s] * sC[s];
    }
    const float zv = xz[rl * (2 * D_INNER) + D_INNER + d];
    y[rl * D_INNER + d] = (acc + Dv * xv) * siluf(zv);
    __syncthreads();
  }
}

// ---------------------------------------------------------------------------
// Attention pooling: scores = (enc@Wk + bk)·q, mask, softmax over L,
// pooled = sum_l w_l * enc.  keys precomputed by GEMM.  One block per batch.
// ---------------------------------------------------------------------------
__global__ void PNI_attn_pool(const float* __restrict__ enc,
                              const float* __restrict__ keys,
                              const float* __restrict__ raw, int cin,
                              const float* __restrict__ q,
                              float* __restrict__ pooled,
                              float* __restrict__ site, int L)
{
  __shared__ float sc[LPEP];
  __shared__ float red[256];
  const int b = blockIdx.x;
  const int t = threadIdx.x;

  for (int l = t; l < L; l += 256) {
    bool m = false;
    const float* rp = raw + ((long)(b * L + l)) * cin;
    for (int c = 0; c < cin; ++c) m = m || (rp[c] != 0.f);
    float s = -3.0e38f;
    if (m) {
      s = 0.f;
      const float* kp = keys + ((long)(b * L + l)) * D_MODEL;
      for (int dd = 0; dd < D_MODEL; ++dd) s += kp[dd] * q[dd];
    }
    sc[l] = s;
  }
  __syncthreads();

  float pm = -3.0e38f;
  for (int l = t; l < L; l += 256) pm = fmaxf(pm, sc[l]);
  red[t] = pm; __syncthreads();
  for (int off = 128; off > 0; off >>= 1) {
    if (t < off) red[t] = fmaxf(red[t], red[t + off]);
    __syncthreads();
  }
  const float mx = red[0]; __syncthreads();

  float ps = 0.f;
  for (int l = t; l < L; l += 256) { float e = __expf(sc[l] - mx); sc[l] = e; ps += e; }
  red[t] = ps; __syncthreads();
  for (int off = 128; off > 0; off >>= 1) {
    if (t < off) red[t] += red[t + off];
    __syncthreads();
  }
  const float inv = 1.f / red[0]; __syncthreads();

  for (int l = t; l < L; l += 256) {
    sc[l] *= inv;
    if (site) site[(long)b * L + l] = sc[l];
  }
  __syncthreads();

  float acc = 0.f;
  for (int l = 0; l < L; ++l)
    acc += sc[l] * enc[((long)(b * L + l)) * D_MODEL + t];
  pooled[b * D_MODEL + t] = acc;
}

// ---------------------------------------------------------------------------
// MLP head: relu([pp,pl] @ fc1^T + b1) @ fc2^T + b2  -> out (B,2)
// ---------------------------------------------------------------------------
__global__ void PNI_head(const float* __restrict__ pp,
                         const float* __restrict__ pl,
                         const float* __restrict__ w1,
                         const float* __restrict__ b1,
                         const float* __restrict__ w2,
                         const float* __restrict__ b2,
                         float* __restrict__ out)
{
  __shared__ float h[D_MODEL];
  const int b = blockIdx.x;
  const int t = threadIdx.x;
  float acc = b1[t];
  const float* w1r = w1 + (long)t * (2 * D_MODEL);
  for (int k = 0; k < D_MODEL; ++k) acc += pp[b * D_MODEL + k] * w1r[k];
  for (int k = 0; k < D_MODEL; ++k) acc += pl[b * D_MODEL + k] * w1r[D_MODEL + k];
  h[t] = fmaxf(acc, 0.f);
  __syncthreads();
  if (t < 2) {
    float o = b2[t];
    for (int dd = 0; dd < D_MODEL; ++dd) o += h[dd] * w2[t * D_MODEL + dd];
    out[b * 2 + t] = o;
  }
}

// ---------------------------------------------------------------------------
static inline void launch_gemm(const float* A, int lda, const float* W,
                               const float* bias, float* C,
                               int M, int N, int K, int act, hipStream_t s)
{
  dim3 grid(M / 16, (N + 255) / 256);
  hipLaunchKernelGGL(PNI_gemm_wmma, grid, dim3(128), 0, s,
                     A, lda, W, bias, C, M, N, K, act);
}

struct PathW {
  const float *emb_w, *emb_b, *in_w, *conv_w, *conv_b, *xproj_w,
              *dt_w, *dt_b, *A_log, *D, *out_w, *q, *key_w, *key_b;
};

static void run_path(const float* raw, int cin, int L, const PathW& w,
                     float* emb, float* xz, float* xc, float* dbl, float* dt,
                     float* mout, float* keys,
                     float* pooled, float* site, hipStream_t s)
{
  const int M = BATCH * L;
  // embed: (B,L,cin) @ emb_w(256,cin)^T + b   (K = 22/5 -> tail-only path)
  launch_gemm(raw, cin, w.emb_w, w.emb_b, emb, M, D_MODEL, cin, 0, s);
  // in_proj: (M,256) @ in_w(1024,256)^T
  launch_gemm(emb, D_MODEL, w.in_w, nullptr, xz, M, 2 * D_INNER, D_MODEL, 0, s);
  // depthwise conv + silu
  {
    int total = M * D_INNER;
    hipLaunchKernelGGL(PNI_conv_silu, dim3((total + 255) / 256), dim3(256), 0, s,
                       xz, w.conv_w, w.conv_b, xc, L);
  }
  // x_proj: (M,512) @ xproj_w(48,512)^T
  launch_gemm(xc, D_INNER, w.xproj_w, nullptr, dbl, M, DT_RANK + 2 * D_STATE,
              D_INNER, 0, s);
  // dt_proj + softplus: dbl[:, :16] @ dt_w(512,16)^T + dt_b  (K=16 tail-only)
  launch_gemm(dbl, DT_RANK + 2 * D_STATE, w.dt_w, w.dt_b, dt, M, D_INNER,
              DT_RANK, 1, s);
  // fused selective scan (+D skip, +silu(z) gate); y written in place over xc
  hipLaunchKernelGGL(PNI_scan, dim3(BATCH), dim3(D_INNER), 0, s,
                     dbl, dt, xc, xz, w.A_log, w.D, xc, L);
  // out_proj
  launch_gemm(xc, D_INNER, w.out_w, nullptr, mout, M, D_MODEL, D_INNER, 0, s);
  // keys = mout @ key_w^T + key_b
  launch_gemm(mout, D_MODEL, w.key_w, w.key_b, keys, M, D_MODEL, D_MODEL, 0, s);
  // masked softmax pooling
  hipLaunchKernelGGL(PNI_attn_pool, dim3(BATCH), dim3(256), 0, s,
                     mout, keys, raw, cin, w.q, pooled, site, L);
}

extern "C" void kernel_launch(void* const* d_in, const int* in_sizes, int n_in,
                              void* d_out, int out_size, void* d_ws, size_t ws_size,
                              hipStream_t stream)
{
  (void)in_sizes; (void)n_in; (void)out_size; (void)ws_size;
  const float* peptide = (const float*)d_in[0];
  const float* ligand  = (const float*)d_in[1];
  PathW pep, lig;
  const float** pw = (const float**)&pep;
  const float** lw = (const float**)&lig;
  for (int i = 0; i < 14; ++i) { pw[i] = (const float*)d_in[2 + i];
                                 lw[i] = (const float*)d_in[16 + i]; }
  const float* fc1_w = (const float*)d_in[30];
  const float* fc1_b = (const float*)d_in[31];
  const float* fc2_w = (const float*)d_in[32];
  const float* fc2_b = (const float*)d_in[33];
  float* out = (float*)d_out;

  // workspace layout (floats), sized for the peptide path and reused for lig
  float* ws    = (float*)d_ws;
  float* emb   = ws;                  // 16*512*256   = 2,097,152
  float* xz    = emb  + 2097152;      // 16*512*1024  = 8,388,608
  float* xc    = xz   + 8388608;      // 16*512*512   = 4,194,304 (also y)
  float* dbl   = xc   + 4194304;      // 16*512*48    =   393,216
  float* dt    = dbl  + 393216;       // 16*512*512   = 4,194,304
  float* mout  = dt   + 4194304;      // 16*512*256   = 2,097,152
  float* keys  = mout + 2097152;      // 16*512*256   = 2,097,152
  float* ppool = keys + 2097152;      // 16*256       =     4,096
  float* lpool = ppool + 4096;        // 16*256       =     4,096

  // output layout: out (B,2) = 32 floats, then site (B,LP) = 8192 floats
  float* site = out + BATCH * 2;

  run_path(peptide, 22, LPEP, pep, emb, xz, xc, dbl, dt, mout, keys,
           ppool, site, stream);
  run_path(ligand, 5, LLIG, lig, emb, xz, xc, dbl, dt, mout, keys,
           lpool, nullptr, stream);

  hipLaunchKernelGGL(PNI_head, dim3(BATCH), dim3(256), 0, stream,
                     ppool, lpool, fc1_w, fc1_b, fc2_w, fc2_b, out);
}
